// DCNV2Fusion_26259430048584
// MI455X (gfx1250) — matrix operand order repaired and checked by reading, not compiled, not run on hardware
//
#include <hip/hip_runtime.h>
#include <hip/hip_bf16.h>
#include <math.h>

// ---------------------------------------------------------------------------
// Problem constants (from reference: B=2, H=W=188, FEAT_CH=OFFSET_CH=OUT_CH=64)
// ---------------------------------------------------------------------------
#define KKK 9
#define CO 64
#define QD 16
#define HH 188
#define WWD 188
#define HWSZ (HH * WWD)          // 35344
#define BB 2
#define NPOS (BB * HWSZ)         // 70688  (divisible by 16)
#define TILE 16
#define NTILES (NPOS / TILE)     // 4418

typedef __attribute__((ext_vector_type(16))) _Float16 v16h;
typedef __attribute__((ext_vector_type(8)))  float    v8f;
typedef __attribute__((ext_vector_type(4)))  unsigned int u32x4;
typedef __attribute__((ext_vector_type(8)))  int      i32x8;
typedef __attribute__((ext_vector_type(4)))  int      i32x4;

union Frag16 { v16h v; _Float16 h[16]; unsigned int u[8]; };

// ---------------------------------------------------------------------------
// TDM: DMA a 2D tile of f32 (width x height, row stride in elements) from
// global memory into LDS. Descriptor built per CDNA5 ISA.
// Groups 2/3 zero (2D tensor). Arity differs between toolchains.
// ---------------------------------------------------------------------------
__device__ __forceinline__ void tdm_load_2d_f32(unsigned lds_byte_addr,
                                                const void* gptr,
                                                unsigned width, unsigned height,
                                                unsigned stride) {
    unsigned long long ga = (unsigned long long)(uintptr_t)gptr;
    u32x4 g0;
    g0[0] = 1u;                                  // count=1 (valid), user mode
    g0[1] = lds_byte_addr;                       // lds_addr [63:32]
    g0[2] = (unsigned)(ga & 0xFFFFFFFFu);        // global_addr lo
    g0[3] = (unsigned)((ga >> 32) & 0x01FFFFFFu) // global_addr [56:32]
          | (2u << 30);                          // type=2 ("image")
    i32x8 g1;
    g1[0] = (int)(2u << 16);                     // workgroup_mask=0, data_size=2 (4B)
    g1[1] = (int)((width & 0xFFFFu) << 16);      // tensor_dim0[15:0] @ bits63:48
    g1[2] = (int)(((height & 0xFFFFu) << 16) |   // tensor_dim1[15:0] @ bits95:80
                  ((width >> 16) & 0xFFFFu));    // tensor_dim0[31:16]
    g1[3] = (int)((width & 0xFFFFu) << 16);      // tile_dim0 @ bits127:112 (dim1 hi = 0)
    g1[4] = (int)(height & 0xFFFFu);             // tile_dim1 @ bits143:128, tile_dim2=0
    g1[5] = (int)stride;                         // tensor_dim0_stride[31:0]
    g1[6] = 0;                                   // stride hi / dim1_stride lo
    g1[7] = 0;
    i32x4 z4 = {0, 0, 0, 0};
#if __clang_major__ >= 23
    i32x8 z8 = {0, 0, 0, 0, 0, 0, 0, 0};
    __builtin_amdgcn_tensor_load_to_lds(g0, g1, z4, z4, z8, 0);
#else
    __builtin_amdgcn_tensor_load_to_lds(g0, g1, z4, z4, 0);
#endif
}

// ---------------------------------------------------------------------------
// Prep: convert deform weights (Co,Ci,3,3) f32 -> f16 [Co][tap*Cpad + ci],
// zero-padded in ci so the WMMA K loop runs in clean chunks of 32.
// ---------------------------------------------------------------------------
__global__ void pack_weights_f16(const float* __restrict__ w,
                                 _Float16* __restrict__ w2,
                                 int Ci, int Cpad) {
    int i = blockIdx.x * blockDim.x + threadIdx.x;
    int total = CO * KKK * Cpad;
    if (i >= total) return;
    int ci  = i % Cpad;
    int tap = (i / Cpad) % KKK;
    int o   = i / (Cpad * KKK);
    float val = (ci < Ci) ? w[((size_t)o * Ci + ci) * KKK + tap] : 0.0f;
    w2[(size_t)o * (KKK * Cpad) + (size_t)tap * Cpad + ci] = (_Float16)val;
}

// ---------------------------------------------------------------------------
// Main deformable-conv kernel: one wave32 per tile of 16 spatial positions.
//   M (16x4 tiles) = output channels, N = 16 positions, K = tap*Cpad + ci.
//   A = f16 weights, B = f16 bilinear-sampled & mask-modulated activations,
//   accumulate in f32 via v_wmma_f32_16x16x32_f16.
// ---------------------------------------------------------------------------
__global__ __launch_bounds__(32)
void deform_wmma(const float* __restrict__ x,      // (B,Ci,H,W) branch features
                 const float* __restrict__ orig,   // (B,64,H,W) offset source
                 const float* __restrict__ offw,   // (27,64)
                 const float* __restrict__ offb,   // (27)
                 const _Float16* __restrict__ w2,  // (64, KKK*Cpad) packed f16
                 const float* __restrict__ bias,   // (64)
                 float* __restrict__ outF,         // (B,64,H,W)
                 int Ci, int Cpad)
{
    __shared__ float raw [TILE][3 * KKK];
    __shared__ float s_py[TILE][KKK];
    __shared__ float s_px[TILE][KKK];
    __shared__ float s_m [TILE][KKK];

    const int lane = threadIdx.x;
    const int pos0 = blockIdx.x * TILE;

    // ---- stage 1: 1x1 offset conv, 16 positions x 27 outputs, cooperative ----
    for (int idx = lane; idx < TILE * 27; idx += 32) {
        int p = idx / 27, o = idx % 27;
        int pos = pos0 + p;
        int b   = (pos >= HWSZ) ? 1 : 0;
        int rem = pos - b * HWSZ;
        const float* op = orig + (size_t)b * CO * HWSZ + rem;
        const float* wr = offw + o * CO;
        float acc = offb[o];
        #pragma unroll 8
        for (int c = 0; c < CO; ++c) acc += wr[c] * op[(size_t)c * HWSZ];
        raw[p][o] = acc;
    }
    __syncthreads();

    // ---- stage 2: sample coordinates (py,px) + sigmoid mask per (pos,tap) ----
    for (int idx = lane; idx < TILE * KKK; idx += 32) {
        int p = idx / KKK, t = idx % KKK;
        int pos = pos0 + p;
        int b   = (pos >= HWSZ) ? 1 : 0;
        int rem = pos - b * HWSZ;
        int yy = rem / WWD, xx = rem % WWD;
        float dy = raw[p][t], dx = raw[p][t + KKK], mm = raw[p][t + 2 * KKK];
        s_py[p][t] = dy + (float)(yy - 1) + (float)(t / 3);
        s_px[p][t] = dx + (float)(xx - 1) + (float)(t % 3);
        s_m [p][t] = 1.0f / (1.0f + __expf(-mm));
    }
    __syncthreads();

    // ---- stage 3: gather -> pack f16 fragments -> WMMA accumulate ----
    const int p  = lane & 15;   // N index (position in tile) / M row for A
    const int hi = lane >> 4;   // lane half selects contiguous K half: k = 16*hi + element
    const int pos = pos0 + p;
    const int b   = (pos >= HWSZ) ? 1 : 0;
    const int rem = pos - b * HWSZ;
    const float* xb = x + (size_t)b * Ci * HWSZ;
    const int nchunk = Cpad >> 5;

    v8f acc[4] = {};   // 4 M-tiles of 16x16 f32 accumulators (Co = 64)

    for (int t = 0; t < KKK; ++t) {
        // bilinear setup for this lane's position at tap t
        float py = s_py[p][t], px = s_px[p][t], mm = s_m[p][t];
        float fy0 = floorf(py), fx0 = floorf(px);
        int   y0 = (int)fy0,    x0 = (int)fx0;
        float wy = py - fy0,    wx = px - fx0;
        int   y1 = y0 + 1,      x1 = x0 + 1;
        float v00 = (y0 >= 0 && y0 < HH && x0 >= 0 && x0 < WWD) ? 1.0f : 0.0f;
        float v01 = (y0 >= 0 && y0 < HH && x1 >= 0 && x1 < WWD) ? 1.0f : 0.0f;
        float v10 = (y1 >= 0 && y1 < HH && x0 >= 0 && x0 < WWD) ? 1.0f : 0.0f;
        float v11 = (y1 >= 0 && y1 < HH && x1 >= 0 && x1 < WWD) ? 1.0f : 0.0f;
        int yc0 = min(max(y0, 0), HH - 1),  yc1 = min(max(y1, 0), HH - 1);
        int xc0 = min(max(x0, 0), WWD - 1), xc1 = min(max(x1, 0), WWD - 1);
        int i00 = yc0 * WWD + xc0, i01 = yc0 * WWD + xc1;
        int i10 = yc1 * WWD + xc0, i11 = yc1 * WWD + xc1;
        float c00 = mm * (1.0f - wy) * (1.0f - wx) * v00;
        float c01 = mm * (1.0f - wy) * wx          * v01;
        float c10 = mm * wy          * (1.0f - wx) * v10;
        float c11 = mm * wy          * wx          * v11;

        for (int ch = 0; ch < nchunk; ++ch) {
            // B fragment (16-bit, 32x16): lanes 0-15 -> K 2v,2v+1; 16-31 -> 16+2v,17+2v.
            // Each lane packs 16 CONSECUTIVE channels [cbase, cbase+16).
            const int cbase = (ch << 5) + hi * 16;
            __builtin_prefetch(xb + (size_t)(cbase + 32) * HWSZ + i00, 0, 1);
            Frag16 bf;
            #pragma unroll
            for (int e = 0; e < 16; ++e) {
                int ci  = cbase + e;
                int cic = (ci < Ci) ? ci : (Ci - 1);      // branchless tail clamp
                const float* cp = xb + (size_t)cic * HWSZ;
                float val = c00 * cp[i00] + c01 * cp[i01]
                          + c10 * cp[i10] + c11 * cp[i11];
                bf.h[e] = (_Float16)((ci < Ci) ? val : 0.0f);
            }
            // A fragment (16-bit, 16x32): lanes 0-15 -> K {0..15}; 16-31 -> {16..31}
            const size_t krow = (size_t)t * Cpad + (ch << 5);
            #pragma unroll
            for (int mt = 0; mt < 4; ++mt) {
                const int o = mt * 16 + p;   // M = lane&15 in both halves
                const unsigned int* wr =
                    (const unsigned int*)(w2 + (size_t)o * (KKK * Cpad) + krow);
                Frag16 af;
                #pragma unroll
                for (int v = 0; v < 8; ++v) {
                    int klo = hi * 16 + 2 * v;
                    af.u[v] = wr[klo >> 1];   // two consecutive f16 K-values
                }
                acc[mt] = __builtin_amdgcn_wmma_f32_16x16x32_f16(
                    false, af.v, false, bf.v, (short)0, acc[mt], false, false);
            }
        }
    }

    // ---- stage 4: bias + ReLU + store per the f32 16x16 C/D layout ----
    // VGPR r: lanes 0-15 -> M=r, N=lane; lanes 16-31 -> M=r+8, N=lane-16
    #pragma unroll
    for (int mt = 0; mt < 4; ++mt) {
        #pragma unroll
        for (int r = 0; r < 8; ++r) {
            int o = mt * 16 + hi * 8 + r;
            float val = acc[mt][r] + bias[o];
            outF[((size_t)b * CO + o) * HWSZ + rem] = fmaxf(val, 0.0f);
        }
    }
}

// ---------------------------------------------------------------------------
// Fusion: key blocks (1x1 conv 64->16 + BN + ReLU), q.k attention, softmax,
// weighted blend. The four 16x64 weight matrices are broadcast operands for
// the whole workgroup -> DMA them into LDS once via the Tensor Data Mover.
// ---------------------------------------------------------------------------
struct KBP { const float *w, *b, *g, *beta, *m, *v; };

__device__ __forceinline__ void key16_lds(const float* __restrict__ x, int b, int rem,
                                          const float* __restrict__ wlds,
                                          const KBP& pr, float* kv) {
    float t[QD];
    #pragma unroll
    for (int j = 0; j < QD; ++j) t[j] = pr.b[j];
    for (int c = 0; c < CO; ++c) {
        float xv = x[((size_t)b * CO + c) * HWSZ + rem];
        #pragma unroll
        for (int j = 0; j < QD; ++j) t[j] += wlds[j * CO + c] * xv;
    }
    #pragma unroll
    for (int j = 0; j < QD; ++j) {
        float s = pr.g[j] * rsqrtf(pr.v[j] + 1e-5f);
        kv[j] = fmaxf(t[j] * s + pr.beta[j] - pr.m[j] * s, 0.0f);
    }
}

__global__ __launch_bounds__(256)
void fuse_attention(const float* __restrict__ orig,
                    const float* __restrict__ fA,
                    const float* __restrict__ fB,
                    const float* __restrict__ fC,
                    KBP q, KBP kA, KBP kB, KBP kC,
                    float* __restrict__ out) {
    __shared__ float sW[4][QD * CO];   // 16 KB: q, kA, kB, kC weights

    // Wave 0 DMAs all four weight matrices into LDS via TDM, then waits on
    // TENSORcnt; the workgroup barrier publishes LDS to the other waves.
    if ((threadIdx.x >> 5) == 0) {
        unsigned lds0 = (unsigned)(uintptr_t)&sW[0][0];
        tdm_load_2d_f32(lds0,                          q.w,  CO, QD, CO);
        tdm_load_2d_f32(lds0 + 1 * QD * CO * 4,        kA.w, CO, QD, CO);
        tdm_load_2d_f32(lds0 + 2 * QD * CO * 4,        kB.w, CO, QD, CO);
        tdm_load_2d_f32(lds0 + 3 * QD * CO * 4,        kC.w, CO, QD, CO);
        __builtin_amdgcn_s_wait_tensorcnt(0);
    }
    __syncthreads();

    int pos = blockIdx.x * blockDim.x + threadIdx.x;
    if (pos >= NPOS) return;
    int b   = (pos >= HWSZ) ? 1 : 0;
    int rem = pos - b * HWSZ;

    float qv[QD], kv[QD];
    key16_lds(orig, b, rem, sW[0], q, qv);

    float a[3];
    key16_lds(fA, b, rem, sW[1], kA, kv);
    { float d = 0.f; for (int j = 0; j < QD; ++j) d += qv[j] * kv[j]; a[0] = d; }
    key16_lds(fB, b, rem, sW[2], kB, kv);
    { float d = 0.f; for (int j = 0; j < QD; ++j) d += qv[j] * kv[j]; a[1] = d; }
    key16_lds(fC, b, rem, sW[3], kC, kv);
    { float d = 0.f; for (int j = 0; j < QD; ++j) d += qv[j] * kv[j]; a[2] = d; }

    float mx = fmaxf(a[0], fmaxf(a[1], a[2]));
    float e0 = __expf(a[0] - mx), e1 = __expf(a[1] - mx), e2 = __expf(a[2] - mx);
    float inv = 1.0f / (e0 + e1 + e2);
    e0 *= inv; e1 *= inv; e2 *= inv;

    for (int o = 0; o < CO; ++o) {
        size_t idx = ((size_t)b * CO + o) * HWSZ + rem;
        out[idx] = e0 * fA[idx] + e1 * fB[idx] + e2 * fC[idx];
    }
}

// ---------------------------------------------------------------------------
// Launcher.  Input order (flattened setup_inputs dict):
//  0 orig_feat  1 center_feat  2 corner_feat  3 foreground_feat
//  4..7   ctr: off_w off_b w b
//  8..11  cnr: off_w off_b w b
// 12..15  fg : off_w off_b w b
// 16..21  q : w b g beta m v   22..27 ctr_k   28..33 cnr_k   34..39 fg_k
// ---------------------------------------------------------------------------
extern "C" void kernel_launch(void* const* d_in, const int* in_sizes, int n_in,
                              void* d_out, int out_size, void* d_ws, size_t ws_size,
                              hipStream_t stream) {
    (void)in_sizes; (void)n_in; (void)out_size; (void)ws_size;
    const float* orig = (const float*)d_in[0];
    const float* ctrF = (const float*)d_in[1];
    const float* cnrF = (const float*)d_in[2];
    const float* fgF  = (const float*)d_in[3];

    const int CI_CTR = 74,  CP_CTR = 96;
    const int CI_CNR = 104, CP_CNR = 128;
    const int CI_FG  = 74,  CP_FG  = 96;

    // workspace carve-up
    char* ws = (char*)d_ws;
    size_t off = 0;
    auto alignup = [](size_t x) { return (x + 255) & ~(size_t)255; };
    _Float16* w2_ctr = (_Float16*)(ws + off); off = alignup(off + (size_t)CO * KKK * CP_CTR * 2);
    _Float16* w2_cnr = (_Float16*)(ws + off); off = alignup(off + (size_t)CO * KKK * CP_CNR * 2);
    _Float16* w2_fg  = (_Float16*)(ws + off); off = alignup(off + (size_t)CO * KKK * CP_FG  * 2);
    const size_t featN = (size_t)BB * CO * HWSZ;
    float* feat0 = (float*)(ws + off); off = alignup(off + featN * 4);
    float* feat1 = (float*)(ws + off); off = alignup(off + featN * 4);
    float* feat2 = (float*)(ws + off); off = alignup(off + featN * 4);

    // 1) pack deform weights to f16 (padded K layout)
    {
        int n0 = CO * KKK * CP_CTR, n1 = CO * KKK * CP_CNR, n2 = CO * KKK * CP_FG;
        pack_weights_f16<<<(n0 + 255) / 256, 256, 0, stream>>>((const float*)d_in[6],  w2_ctr, CI_CTR, CP_CTR);
        pack_weights_f16<<<(n1 + 255) / 256, 256, 0, stream>>>((const float*)d_in[10], w2_cnr, CI_CNR, CP_CNR);
        pack_weights_f16<<<(n2 + 255) / 256, 256, 0, stream>>>((const float*)d_in[14], w2_fg,  CI_FG,  CP_FG);
    }

    // 2) three deformable-conv branches (WMMA core)
    deform_wmma<<<NTILES, 32, 0, stream>>>(ctrF, orig,
        (const float*)d_in[4],  (const float*)d_in[5],  w2_ctr, (const float*)d_in[7],
        feat0, CI_CTR, CP_CTR);
    deform_wmma<<<NTILES, 32, 0, stream>>>(cnrF, orig,
        (const float*)d_in[8],  (const float*)d_in[9],  w2_cnr, (const float*)d_in[11],
        feat1, CI_CNR, CP_CNR);
    deform_wmma<<<NTILES, 32, 0, stream>>>(fgF, orig,
        (const float*)d_in[12], (const float*)d_in[13], w2_fg,  (const float*)d_in[15],
        feat2, CI_FG, CP_FG);

    // 3) key blocks + attention fusion (TDM-staged weights)
    KBP q  = { (const float*)d_in[16], (const float*)d_in[17], (const float*)d_in[18],
               (const float*)d_in[19], (const float*)d_in[20], (const float*)d_in[21] };
    KBP kA = { (const float*)d_in[22], (const float*)d_in[23], (const float*)d_in[24],
               (const float*)d_in[25], (const float*)d_in[26], (const float*)d_in[27] };
    KBP kB = { (const float*)d_in[28], (const float*)d_in[29], (const float*)d_in[30],
               (const float*)d_in[31], (const float*)d_in[32], (const float*)d_in[33] };
    KBP kC = { (const float*)d_in[34], (const float*)d_in[35], (const float*)d_in[36],
               (const float*)d_in[37], (const float*)d_in[38], (const float*)d_in[39] };
    fuse_attention<<<(NPOS + 255) / 256, 256, 0, stream>>>(orig, feat0, feat1, feat2,
                                                           q, kA, kB, kC, (float*)d_out);
}
